// VectorQuantizer_62405874811226
// MI455X (gfx1250) — matrix operand order repaired and checked by reading, not compiled, not run on hardware
//
#include <hip/hip_runtime.h>
#include <cstdint>

// ---------------------------------------------------------------------------
// VQ nearest-codebook search on gfx1250 via split-precision bf16 WMMA.
//   dist ranking:  e_sq[k] - 2 * (z . e_k)   (z_sq constant per row)
//   z.e computed as zh*eh + zh*el + zl*eh  (bf16 hi/lo split, ~fp32 accuracy)
// Round 3: dual accumulator chains (2 column tiles/stage) + async-to-LDS
// double buffering (ASYNCcnt) so global latency hides behind WMMA compute.
// ---------------------------------------------------------------------------

typedef __attribute__((ext_vector_type(16))) __bf16 v16bf;
typedef __attribute__((ext_vector_type(8)))  float  v8f;
typedef __attribute__((ext_vector_type(4)))  int    v4i;

#define N_ROWS   65536            // N_VECS
#define K_VECS   8192             // NUM_TOKENS (codebook size)
#define D_DIM    256              // EMBED_DIM
#define KSTEPS   8                // D_DIM / 32
#define NTILES   (K_VECS / 16)    // 512 column tiles (16 cols each)
#define NSTAGES  (K_VECS / 32)    // 256 stages (2 tiles each)
#define TILE_Q   512              // uint4 per 16-col tile per (hi|lo) buffer
#define STAGE_Q  1024             // uint4 per stage per (hi|lo) buffer
#define BETA_F   0.25f

union Frag16 { v16bf v; uint4 q[2]; };
union Frag8  { __bf16 b[8]; uint4 q; };

#if __has_builtin(__builtin_amdgcn_global_load_async_to_lds_b128)
#define ASYNC_CP 1
typedef __attribute__((address_space(1))) v4i* gptr_v4i;
typedef __attribute__((address_space(3))) v4i* lptr_v4i;
#endif

// Copy 16 bytes global -> LDS. Async path uses the CDNA5 async-to-LDS DMA
// (tracked by ASYNCcnt); fallback is a synchronous load+ds_store.
__device__ __forceinline__ void cp16(uint4* l, const uint4* g) {
#ifdef ASYNC_CP
    __builtin_amdgcn_global_load_async_to_lds_b128(
        (gptr_v4i)g, (lptr_v4i)l, 0, 0);
#else
    *l = *g;
#endif
}

__device__ __forceinline__ void async_wait() {
#ifdef ASYNC_CP
#if __has_builtin(__builtin_amdgcn_s_wait_asynccnt)
    __builtin_amdgcn_s_wait_asynccnt(0);
#else
    asm volatile("s_wait_asynccnt 0x0" ::: "memory");
#endif
#endif
}

// ---------------------------------------------------------------------------
// Kernel 1: repack embeddings into WMMA B-fragment layout, bf16 hi + lo.
// B (32x16, 16-bit): col n = lane&15; lanes 0-15 hold K=0..7,16..23, lanes 16-31
// hold K=8..15,24..31; VGPR chunk c (16B) = 8 consecutive K values.
// Element e[k][d]:  t=k>>4, n=k&15, s=d>>5, h=(d>>3)&1, c=(d>>4)&1
//   uint4 index = (t*16 + s*2 + c)*32 + h*16 + n
// One thread handles 8 consecutive d for one k.
// ---------------------------------------------------------------------------
__global__ __launch_bounds__(256) void k_pack(const float* __restrict__ e,
                                              uint4* __restrict__ Bhi,
                                              uint4* __restrict__ Blo) {
    int id = blockIdx.x * 256 + threadIdx.x;   // 0 .. K_VECS*32-1
    int k  = id >> 5;
    int r  = id & 31;
    int s  = r >> 2;
    int h  = (r >> 1) & 1;
    int c  = r & 1;
    int d0 = s * 32 + c * 16 + h * 8;

    const float* ep = e + (size_t)k * D_DIM + d0;
    float4 x0 = *(const float4*)(ep);
    float4 x1 = *(const float4*)(ep + 4);
    float xs[8] = {x0.x, x0.y, x0.z, x0.w, x1.x, x1.y, x1.z, x1.w};

    Frag8 uh, ul;
#pragma unroll
    for (int j = 0; j < 8; ++j) {
        float x   = xs[j];
        __bf16 hb = (__bf16)x;
        uh.b[j]   = hb;
        ul.b[j]   = (__bf16)(x - (float)hb);
    }
    int t = k >> 4, n = k & 15;
    int qidx = (t * 16 + s * 2 + c) * 32 + h * 16 + n;
    Bhi[qidx] = uh.q;
    Blo[qidx] = ul.q;
}

// ---------------------------------------------------------------------------
// Kernel 2: ||e_k||^2
// ---------------------------------------------------------------------------
__global__ __launch_bounds__(256) void k_esq(const float* __restrict__ e,
                                             float* __restrict__ esq) {
    int k = blockIdx.x * 256 + threadIdx.x;    // 0 .. K_VECS-1
    const float4* p = (const float4*)(e + (size_t)k * D_DIM);
    float s = 0.0f;
#pragma unroll 4
    for (int i = 0; i < D_DIM / 4; ++i) {
        float4 v = p[i];
        s += v.x * v.x + v.y * v.y + v.z * v.z + v.w * v.w;
    }
    esq[k] = s;
}

// ---------------------------------------------------------------------------
// Kernel 3: fused GEMM + argmin.
// 512 blocks x 256 threads (8 waves). Wave w owns 16 z-rows; its A-fragments
// (hi + lo, 8 K-steps) stay resident in 128 VGPRs for the entire K sweep.
// Each stage covers 32 codebook columns (2 WMMA tiles, independent
// accumulator chains acc0/acc1). LDS is double-buffered: stage st+1 is
// copied global->LDS via async-to-LDS while stage st computes.
// A (16x32, 16-bit) layout: lane<16 -> K in [0,8)U[16,24); lane>=16 ->
// K in [8,16)U[24,32) of row (lane&15).
// C/D layout: VGPR v -> M = v + 8*(lane>>4), N = lane&15.
// ---------------------------------------------------------------------------
__global__ __launch_bounds__(256) void k_argmin(const float* __restrict__ z,
                                                const uint4* __restrict__ Bhi,
                                                const uint4* __restrict__ Blo,
                                                const float* __restrict__ esq,
                                                int* __restrict__ nearest) {
    __shared__ uint4 sH[2][STAGE_Q];   // 2 x 16 KB
    __shared__ uint4 sL[2][STAGE_Q];   // 2 x 16 KB

    const int tid     = threadIdx.x;
    const int wave    = tid >> 5;
    const int lane    = tid & 31;
    const int half    = lane >> 4;         // 0 | 1
    const int l16     = lane & 15;
    const int rowBase = blockIdx.x * 128 + wave * 16;
    const int myRow   = rowBase + l16;
    const int kb      = half * 8;

    // ---- load + split-convert the wave's A fragments (registers) ----
    v16bf Ah[KSTEPS], Al[KSTEPS];
#pragma unroll
    for (int s = 0; s < KSTEPS; ++s) {
        const float* zp = z + (size_t)myRow * D_DIM + s * 32 + kb;
        float4 x0 = *(const float4*)(zp);
        float4 x1 = *(const float4*)(zp + 4);
        float4 x2 = *(const float4*)(zp + 16);
        float4 x3 = *(const float4*)(zp + 20);
        float xs[16] = {x0.x, x0.y, x0.z, x0.w, x1.x, x1.y, x1.z, x1.w,
                        x2.x, x2.y, x2.z, x2.w, x3.x, x3.y, x3.z, x3.w};
        union { v16bf v; __bf16 b[16]; } H, L;
#pragma unroll
        for (int j = 0; j < 16; ++j) {
            float  x  = xs[j];
            __bf16 hb = (__bf16)x;
            H.b[j] = hb;
            L.b[j] = (__bf16)(x - (float)hb);
        }
        Ah[s] = H.v;
        Al[s] = L.v;
    }

    float minval[8];
    int   minidx[8];
#pragma unroll
    for (int v = 0; v < 8; ++v) { minval[v] = 3.402823466e38f; minidx[v] = 0; }

    // stage filler: 2048 uint4 per stage, 8 copies per thread
    auto fill = [&](int st, int buf) {
        const uint4* gh = Bhi + (size_t)st * STAGE_Q;
        const uint4* gl = Blo + (size_t)st * STAGE_Q;
#pragma unroll
        for (int p = 0; p < 4; ++p) {
            cp16(&sH[buf][tid + p * 256], gh + tid + p * 256);
            cp16(&sL[buf][tid + p * 256], gl + tid + p * 256);
        }
    };

    // prologue: stage 0 into buffer 0
    fill(0, 0);
    async_wait();
    __syncthreads();

    for (int st = 0; st < NSTAGES; ++st) {
        const int cur = st & 1;
        if (st + 1 < NSTAGES) fill(st + 1, cur ^ 1);   // background copy

        float es0 = esq[st * 32 + l16];
        float es1 = esq[st * 32 + 16 + l16];

        v8f acc0 = {}, acc1 = {};
#pragma unroll
        for (int s = 0; s < KSTEPS; ++s) {
            Frag16 bh0, bl0, bh1, bl1;
            const int i0 = (s * 2 + 0) * 32 + lane;
            const int i1 = (s * 2 + 1) * 32 + lane;
            bh0.q[0] = sH[cur][i0];          bh0.q[1] = sH[cur][i1];
            bl0.q[0] = sL[cur][i0];          bl0.q[1] = sL[cur][i1];
            bh1.q[0] = sH[cur][TILE_Q + i0]; bh1.q[1] = sH[cur][TILE_Q + i1];
            bl1.q[0] = sL[cur][TILE_Q + i0]; bl1.q[1] = sL[cur][TILE_Q + i1];
            // two independent accumulation chains, interleaved issue
            acc0 = __builtin_amdgcn_wmma_f32_16x16x32_bf16(
                false, Ah[s], false, bh0.v, (short)0, acc0, false, false);
            acc1 = __builtin_amdgcn_wmma_f32_16x16x32_bf16(
                false, Ah[s], false, bh1.v, (short)0, acc1, false, false);
            acc0 = __builtin_amdgcn_wmma_f32_16x16x32_bf16(
                false, Ah[s], false, bl0.v, (short)0, acc0, false, false);
            acc1 = __builtin_amdgcn_wmma_f32_16x16x32_bf16(
                false, Ah[s], false, bl1.v, (short)0, acc1, false, false);
            acc0 = __builtin_amdgcn_wmma_f32_16x16x32_bf16(
                false, Al[s], false, bh0.v, (short)0, acc0, false, false);
            acc1 = __builtin_amdgcn_wmma_f32_16x16x32_bf16(
                false, Al[s], false, bh1.v, (short)0, acc1, false, false);
        }

        const int c0 = st * 32 + l16;          // tile 0 column, tile 1 = c0+16
#pragma unroll
        for (int v = 0; v < 8; ++v) {
            float d0 = es0 - 2.0f * acc0[v];
            float d1 = es1 - 2.0f * acc1[v];
            if (d0 < minval[v]) { minval[v] = d0; minidx[v] = c0; }
            if (d1 < minval[v]) { minval[v] = d1; minidx[v] = c0 + 16; }
        }

        async_wait();          // next stage's LDS writes complete (this wave)
        __syncthreads();       // whole block: safe to read buf^1 / refill buf
    }

    // ---- cross-lane argmin: N = lane&15 varies within each 16-lane half ----
#pragma unroll
    for (int v = 0; v < 8; ++v) {
        float mv = minval[v];
        int   mi = minidx[v];
#pragma unroll
        for (int off = 1; off < 16; off <<= 1) {
            float ov = __shfl_xor(mv, off, 32);
            int   oi = __shfl_xor(mi, off, 32);
            if (ov < mv || (ov == mv && oi < mi)) { mv = ov; mi = oi; }
        }
        if (l16 == 0) nearest[rowBase + v + half * 8] = mi;  // lanes 0 and 16
    }
}

// ---------------------------------------------------------------------------
// Kernel 4: gather z_q = emb[nearest], partial loss sums, nearest -> out tail.
// ---------------------------------------------------------------------------
__global__ __launch_bounds__(256) void k_gather(const float* __restrict__ z,
                                                const float* __restrict__ e,
                                                const int* __restrict__ nearest,
                                                float* __restrict__ out,
                                                float* __restrict__ partial) {
    __shared__ float red[256];
    const int tid    = blockIdx.x * 256 + threadIdx.x;
    const int stride = 512 * 256;
    const int NF4    = N_ROWS * (D_DIM / 4);   // float4 elements

    const float4* z4 = (const float4*)z;
    const float4* e4 = (const float4*)e;
    float4*       o4 = (float4*)out;

    float lsum = 0.0f;
    for (int i = tid; i < NF4; i += stride) {
        int n   = i >> 6;                      // row (64 float4 per row)
        int d4  = i & 63;
        int idx = nearest[n];
        float4 q = e4[(size_t)idx * 64 + d4];
        o4[i] = q;
        float4 zz = z4[i];
        float dx = zz.x - q.x, dy = zz.y - q.y;
        float dz = zz.z - q.z, dw = zz.w - q.w;
        lsum += dx * dx + dy * dy + dz * dz + dw * dw;
    }

    // nearest indices (int32) into the output tail, after z_q and loss
    if (tid < N_ROWS) ((int*)out)[(size_t)N_ROWS * D_DIM + 1 + tid] = nearest[tid];

    // deterministic block tree reduction
    red[threadIdx.x] = lsum;
    __syncthreads();
    for (int s = 128; s > 0; s >>= 1) {
        if (threadIdx.x < s) red[threadIdx.x] += red[threadIdx.x + s];
        __syncthreads();
    }
    if (threadIdx.x == 0) partial[blockIdx.x] = red[0];
}

// ---------------------------------------------------------------------------
// Kernel 5: finalize loss (fixed-order sum of 512 partials -> deterministic).
// ---------------------------------------------------------------------------
__global__ void k_final(const float* __restrict__ partial, float* __restrict__ out) {
    if (threadIdx.x == 0 && blockIdx.x == 0) {
        float s = 0.0f;
        for (int i = 0; i < 512; ++i) s += partial[i];
        out[(size_t)N_ROWS * D_DIM] =
            (1.0f + BETA_F) * s / ((float)N_ROWS * (float)D_DIM);
    }
}

// ---------------------------------------------------------------------------
extern "C" void kernel_launch(void* const* d_in, const int* in_sizes, int n_in,
                              void* d_out, int out_size, void* d_ws, size_t ws_size,
                              hipStream_t stream) {
    (void)in_sizes; (void)n_in; (void)out_size; (void)ws_size;

    const float* z   = (const float*)d_in[0];   // [65536, 256]
    const float* emb = (const float*)d_in[1];   // [8192, 256]
    float*       out = (float*)d_out;

    char* ws = (char*)d_ws;
    uint4* Bhi     = (uint4*)(ws);                                   // 4 MB
    uint4* Blo     = (uint4*)(ws + ((size_t)4 << 20));               // 4 MB
    float* esq     = (float*)(ws + ((size_t)8 << 20));               // 32 KB
    int*   nearest = (int*)  (ws + ((size_t)8 << 20) + (32u << 10)); // 256 KB
    float* partial = (float*)(ws + ((size_t)8 << 20) + (32u << 10) + (256u << 10));

    k_pack  <<<K_VECS * 32 / 256, 256, 0, stream>>>(emb, Bhi, Blo);
    k_esq   <<<K_VECS / 256,      256, 0, stream>>>(emb, esq);
    k_argmin<<<N_ROWS / 128,      256, 0, stream>>>(z, Bhi, Blo, esq, nearest);
    k_gather<<<512,               256, 0, stream>>>(z, emb, nearest, out, partial);
    k_final <<<1,                 32,  0, stream>>>(partial, out);
}
